// AttnBlock_68719476921
// MI455X (gfx1250) — matrix operand order repaired and checked by reading, not compiled, not run on hardware
//
#include <hip/hip_runtime.h>
#include <hip/hip_bf16.h>

typedef __attribute__((ext_vector_type(16))) _Float16 v16h;
typedef __attribute__((ext_vector_type(8)))  _Float16 v8h;
typedef __attribute__((ext_vector_type(8)))  float    v8f;

#define C_DIM  512
#define N_DIM  4096
#define BATCH  2
#define GROUPS 32
#define NW     4          // waves per GEMM block
#define MT     4          // 16x16 sub-tiles per wave in M
#define NT     4          // 16x16 sub-tiles per wave in N

// ---------------- f32 -> f16 weight conversion ----------------
__global__ void cvt_f32_f16_kernel(const float* __restrict__ src,
                                   _Float16* __restrict__ dst, int n) {
    int i = blockIdx.x * blockDim.x + threadIdx.x;
    if (i < n) dst[i] = (_Float16)src[i];
}

// ---------------- GroupNorm -> hnt[b][n][c] (f16, transposed) ----------------
__global__ void groupnorm_kernel(const float* __restrict__ x,
                                 const float* __restrict__ gamma,
                                 const float* __restrict__ beta,
                                 _Float16* __restrict__ hnt) {
    const int CPG   = C_DIM / GROUPS;   // 16
    const int ELEMS = CPG * N_DIM;      // 65536
    int b = blockIdx.x / GROUPS;
    int g = blockIdx.x % GROUPS;
    const float* xp = x + ((size_t)b * C_DIM + g * CPG) * N_DIM;

    float s = 0.f, ss = 0.f;
    for (int e = threadIdx.x; e < ELEMS; e += blockDim.x) {
        float v = xp[e]; s += v; ss += v * v;
    }
    for (int o = 16; o; o >>= 1) {
        s  += __shfl_down(s,  o, 32);
        ss += __shfl_down(ss, o, 32);
    }
    __shared__ float red[2][8];
    __shared__ float stats[2];
    int lane = threadIdx.x & 31, wid = threadIdx.x >> 5;
    if (lane == 0) { red[0][wid] = s; red[1][wid] = ss; }
    __syncthreads();
    if (threadIdx.x == 0) {
        float S = 0.f, SS = 0.f;
        for (int i = 0; i < 8; ++i) { S += red[0][i]; SS += red[1][i]; }
        float mean = S / (float)ELEMS;
        float var  = SS / (float)ELEMS - mean * mean;
        stats[0] = mean;
        stats[1] = rsqrtf(var + 1e-6f);
    }
    __syncthreads();
    float mean = stats[0], rstd = stats[1];
    for (int e = threadIdx.x; e < ELEMS; e += blockDim.x) {
        int cl = e >> 12;            // e / 4096
        int n  = e & (N_DIM - 1);
        int c  = g * CPG + cl;
        float v = (xp[e] - mean) * rstd * gamma[c] + beta[c];
        hnt[((size_t)b * N_DIM + n) * C_DIM + c] = (_Float16)v;
    }
}

// ---------------- generic WMMA f16 GEMM, 64x64 macro-tile per wave ----------------
// D[m][n] = sum_k Arow(m)[k] * Brow(n)[k]   (both operands read as rows)
// A-operand ISA layout per lane: halves 0..7  <- K = k0 + 8*hi .. +7,
//                                halves 8..15 <- K = k0 + 16 + 8*hi .. +7
// B-operand per lane: same layout as A
#define OUT_F16  0
#define OUT_F32  1
#define BIAS_NONE 0
#define BIAS_ROW  1
#define BIAS_COL  2

__device__ __forceinline__ v16h load_a16(const _Float16* p) {
    v8h lo = *(const v8h*)p;
    v8h hi = *(const v8h*)(p + 16);
    v16h r;
    #pragma unroll
    for (int e = 0; e < 8; ++e) { r[e] = lo[e]; r[8 + e] = hi[e]; }
    return r;
}

template <int OUTM, int BIASM, int TRANS, int RES>
__global__ void __launch_bounds__(32 * NW)
gemm_wmma(const _Float16* __restrict__ A,  long long sA, int lda,
          const _Float16* __restrict__ Bm, long long sB, int ldb,
          void* __restrict__ Dp,           long long sD, int ldd,
          const float* __restrict__ bias,
          const float* __restrict__ resid, long long sR,
          int K) {
    int lane = threadIdx.x;          // 0..31
    int hi   = lane >> 4;
    int lr   = lane & 15;
    int n0   = (blockIdx.x * NW + threadIdx.y) * (16 * NT);
    int m0   = blockIdx.y * (16 * MT);
    int bz   = blockIdx.z;

    const _Float16* ar[MT];
    const _Float16* br[NT];
    #pragma unroll
    for (int t = 0; t < MT; ++t)
        ar[t] = A + (size_t)bz * sA + (size_t)(m0 + t * 16 + lr) * lda + 8 * hi;
    #pragma unroll
    for (int u = 0; u < NT; ++u)
        br[u] = Bm + (size_t)bz * sB + (size_t)(n0 + u * 16 + lr) * ldb + 8 * hi;

    v8f acc[MT][NT] = {};
    for (int k0 = 0; k0 < K; k0 += 32) {
        v16h a[MT], b[NT];
        #pragma unroll
        for (int t = 0; t < MT; ++t) a[t] = load_a16(ar[t] + k0);
        #pragma unroll
        for (int u = 0; u < NT; ++u) b[u] = load_a16(br[u] + k0);
        #pragma unroll
        for (int t = 0; t < MT; ++t)
            #pragma unroll
            for (int u = 0; u < NT; ++u)
                acc[t][u] = __builtin_amdgcn_wmma_f32_16x16x32_f16(
                    false, a[t], false, b[u], (short)0, acc[t][u], false, false);
    }

    if (TRANS) {
        // store D transposed (f16): lane owns column N, 8 contiguous halves per sub-tile
        #pragma unroll
        for (int t = 0; t < MT; ++t) {
            #pragma unroll
            for (int u = 0; u < NT; ++u) {
                _Float16* d = (_Float16*)Dp + (size_t)bz * sD +
                              (size_t)(n0 + u * 16 + lr) * ldd + m0 + t * 16 + 8 * hi;
                v8h o;
                #pragma unroll
                for (int r = 0; r < 8; ++r) o[r] = (_Float16)acc[t][u][r];
                *(v8h*)d = o;
            }
        }
    } else {
        #pragma unroll
        for (int t = 0; t < MT; ++t) {
            #pragma unroll
            for (int u = 0; u < NT; ++u) {
                int Mrow = m0 + t * 16 + 8 * hi;
                int Ncol = n0 + u * 16 + lr;
                #pragma unroll
                for (int r = 0; r < 8; ++r) {
                    float val = acc[t][u][r];
                    if (BIASM == BIAS_ROW) val += bias[Mrow + r];
                    if (BIASM == BIAS_COL) val += bias[Ncol];
                    size_t idx = (size_t)(Mrow + r) * ldd + Ncol;
                    if (RES) val += resid[(size_t)bz * sR + idx];
                    if (OUTM == OUT_F16)
                        ((_Float16*)Dp)[(size_t)bz * sD + idx] = (_Float16)val;
                    else
                        ((float*)Dp)[(size_t)bz * sD + idx] = val;
                }
            }
        }
    }
}

// ---------------- row softmax: S (f32) -> P (f16), scale = C^-0.5 ----------------
__global__ void softmax_kernel(const float* __restrict__ S,
                               _Float16* __restrict__ P) {
    const float scale = 0.04419417382415922f;   // 1/sqrt(512)
    const float* srow = S + (size_t)blockIdx.x * N_DIM;
    _Float16*    prow = P + (size_t)blockIdx.x * N_DIM;

    float vals[16];
    float m = -3.4e38f;
    #pragma unroll
    for (int i = 0; i < 16; ++i) {
        vals[i] = srow[threadIdx.x + i * 256] * scale;
        m = fmaxf(m, vals[i]);
    }
    for (int o = 16; o; o >>= 1) m = fmaxf(m, __shfl_down(m, o, 32));
    __shared__ float sm[8];
    __shared__ float bc[2];
    int lane = threadIdx.x & 31, wid = threadIdx.x >> 5;
    if (!lane) sm[wid] = m;
    __syncthreads();
    if (threadIdx.x == 0) {
        float mm = sm[0];
        for (int i = 1; i < 8; ++i) mm = fmaxf(mm, sm[i]);
        bc[0] = mm;
    }
    __syncthreads();
    float rowmax = bc[0];
    float s = 0.f;
    #pragma unroll
    for (int i = 0; i < 16; ++i) { vals[i] = __expf(vals[i] - rowmax); s += vals[i]; }
    for (int o = 16; o; o >>= 1) s += __shfl_down(s, o, 32);
    __syncthreads();
    if (!lane) sm[wid] = s;
    __syncthreads();
    if (threadIdx.x == 0) {
        float ss = 0.f;
        for (int i = 0; i < 8; ++i) ss += sm[i];
        bc[1] = 1.0f / ss;
    }
    __syncthreads();
    float inv = bc[1];
    #pragma unroll
    for (int i = 0; i < 16; ++i)
        prow[threadIdx.x + i * 256] = (_Float16)(vals[i] * inv);
}

// ---------------- host launcher ----------------
extern "C" void kernel_launch(void* const* d_in, const int* in_sizes, int n_in,
                              void* d_out, int out_size, void* d_ws, size_t ws_size,
                              hipStream_t stream) {
    (void)in_sizes; (void)n_in; (void)out_size; (void)ws_size;

    const float* x        = (const float*)d_in[0];
    const float* gn_scale = (const float*)d_in[1];
    const float* gn_bias  = (const float*)d_in[2];
    const float* wq = (const float*)d_in[3];
    const float* bq = (const float*)d_in[4];
    const float* wk = (const float*)d_in[5];
    const float* bk = (const float*)d_in[6];
    const float* wv = (const float*)d_in[7];
    const float* bv = (const float*)d_in[8];
    const float* wo = (const float*)d_in[9];
    const float* bo = (const float*)d_in[10];
    float* out = (float*)d_out;

    // workspace carve-up (all offsets 256B aligned)
    char*  ws  = (char*)d_ws;
    size_t off = 0;
    auto alloc = [&](size_t bytes) {
        char* p = ws + off;
        off += (bytes + 255) & ~(size_t)255;
        return p;
    };
    const size_t WW   = (size_t)C_DIM * C_DIM;           // 262144
    const size_t NC   = (size_t)N_DIM * C_DIM;           // 2M elems per batch
    const size_t NN   = (size_t)N_DIM * N_DIM;           // 16M elems per batch
    _Float16* wq_h = (_Float16*)alloc(WW * 2);
    _Float16* wk_h = (_Float16*)alloc(WW * 2);
    _Float16* wv_h = (_Float16*)alloc(WW * 2);
    _Float16* wo_h = (_Float16*)alloc(WW * 2);
    _Float16* hnt  = (_Float16*)alloc(BATCH * NC * 2);   // [b][n][c]
    _Float16* qt   = (_Float16*)alloc(BATCH * NC * 2);   // [b][n][c]
    _Float16* kt   = (_Float16*)alloc(BATCH * NC * 2);   // [b][n][c]
    _Float16* vmat = (_Float16*)alloc(BATCH * NC * 2);   // [b][c][n]
    _Float16* ot   = (_Float16*)alloc(BATCH * NC * 2);   // [b][n][c]
    _Float16* P    = (_Float16*)alloc(BATCH * NN * 2);   // [b][i][j]
    float*    Sb   = (float*)   alloc(BATCH * NN * 4);   // [b][i][j]

    // 1) weights f32 -> f16
    {
        int n = (int)WW, g = (n + 255) / 256;
        cvt_f32_f16_kernel<<<g, 256, 0, stream>>>(wq, wq_h, n);
        cvt_f32_f16_kernel<<<g, 256, 0, stream>>>(wk, wk_h, n);
        cvt_f32_f16_kernel<<<g, 256, 0, stream>>>(wv, wv_h, n);
        cvt_f32_f16_kernel<<<g, 256, 0, stream>>>(wo, wo_h, n);
    }

    // 2) GroupNorm -> hnt[b][n][c]
    groupnorm_kernel<<<BATCH * GROUPS, 256, 0, stream>>>(x, gn_scale, gn_bias, hnt);

    dim3 blk(32, NW);
    long long sNC = (long long)NC, sNN = (long long)NN;
    long long sOUT = (long long)C_DIM * N_DIM;
    const int MTILE = 16 * MT;            // 64
    const int NTILE = 16 * NT * NW;       // 256 per block

    // 3) qt[b][n][o] = sum_i hnt[n][i] * wq[o][i] + bq[o]   (M=N_DIM, N=C_DIM, K=C_DIM)
    gemm_wmma<OUT_F16, BIAS_COL, 0, 0>
        <<<dim3(C_DIM / NTILE, N_DIM / MTILE, BATCH), blk, 0, stream>>>(
            hnt, sNC, C_DIM, wq_h, 0, C_DIM, qt, sNC, C_DIM, bq, nullptr, 0, C_DIM);
    // 4) kt
    gemm_wmma<OUT_F16, BIAS_COL, 0, 0>
        <<<dim3(C_DIM / NTILE, N_DIM / MTILE, BATCH), blk, 0, stream>>>(
            hnt, sNC, C_DIM, wk_h, 0, C_DIM, kt, sNC, C_DIM, bk, nullptr, 0, C_DIM);
    // 5) v[b][o][n] = sum_i wv[o][i] * hnt[n][i] + bv[o]    (M=C_DIM, N=N_DIM, K=C_DIM)
    gemm_wmma<OUT_F16, BIAS_ROW, 0, 0>
        <<<dim3(N_DIM / NTILE, C_DIM / MTILE, BATCH), blk, 0, stream>>>(
            wv_h, 0, C_DIM, hnt, sNC, C_DIM, vmat, sNC, N_DIM, bv, nullptr, 0, C_DIM);
    // 6) S[b][i][j] = sum_c qt[i][c] * kt[j][c]             (M=N=N_DIM, K=C_DIM)
    gemm_wmma<OUT_F32, BIAS_NONE, 0, 0>
        <<<dim3(N_DIM / NTILE, N_DIM / MTILE, BATCH), blk, 0, stream>>>(
            qt, sNC, C_DIM, kt, sNC, C_DIM, Sb, sNN, N_DIM, nullptr, nullptr, 0, C_DIM);
    // 7) softmax rows -> P f16
    softmax_kernel<<<BATCH * N_DIM, 256, 0, stream>>>(Sb, P);
    // 8) O[c][i] = sum_j v[c][j] * P[i][j], stored transposed as ot[i][c]
    //    (M=C_DIM, N=N_DIM, K=N_DIM)
    gemm_wmma<OUT_F16, BIAS_NONE, 1, 0>
        <<<dim3(N_DIM / NTILE, C_DIM / MTILE, BATCH), blk, 0, stream>>>(
            vmat, sNC, N_DIM, P, sNN, N_DIM, ot, sNC, C_DIM, nullptr, nullptr, 0, N_DIM);
    // 9) out[o][i] = sum_c wo[o][c] * ot[i][c] + bo[o] + x  (M=C_DIM, N=N_DIM, K=C_DIM)
    gemm_wmma<OUT_F32, BIAS_ROW, 0, 1>
        <<<dim3(N_DIM / NTILE, C_DIM / MTILE, BATCH), blk, 0, stream>>>(
            wo_h, 0, C_DIM, ot, sNC, C_DIM, out, sOUT, N_DIM, bo, x, sOUT, C_DIM);
}
